// KCGE_77403900608956
// MI455X (gfx1250) — compile-verified, matrix operand order checked by reading, not run on hardware
//
#include <hip/hip_runtime.h>
#include <hip/hip_bf16.h>

// ---------------- CDNA5 (gfx1250, wave32) RGCN: bucketed bf16 WMMA GEMM ----
typedef __attribute__((ext_vector_type(16))) __bf16 v16bf;
typedef __attribute__((ext_vector_type(8)))  float  v8f;

#define E_NODES 200000
#define D_DIM   128
#define R_REL   4
#define NEG_SLOPE 0.01f

// per-weight-set fragment buffer: R * 4(kstep) * 8(nsub) * 32(lane) * 16(elem)
#define WFRAG_SET 65536
#define WFRAG_REL 16384

union Frag16 { v16bf v; uint4 u[2]; };

// A/B 16-bit operand K-mapping (ISA 7.12.2, 16-bit 16x32 A layout):
// lane<16 -> K base 0, lane>=16 -> K base 8; elem j<8 -> +j, j>=8 -> +8+j
__device__ __forceinline__ int kmap(int j, int lane) {
  return ((lane < 16) ? 0 : 8) + ((j < 8) ? j : (8 + j));
}

// ---------------------------------------------------------------- utilities
__global__ void k_zero_f32(float* p, int n) {
  int i = blockIdx.x * blockDim.x + threadIdx.x;
  if (i < n) p[i] = 0.0f;
}
__global__ void k_zero_u32(unsigned* p, int n) {
  int i = blockIdx.x * blockDim.x + threadIdx.x;
  if (i < n) p[i] = 0u;
}
__global__ void k_init_out_x(float* out, __bf16* xbf, int n) {
  int i = blockIdx.x * blockDim.x + threadIdx.x;
  if (i < n) { out[i] = 1.0f; xbf[i] = (__bf16)1.0f; }
}

// ------------------------------------------------------------- degree / norm
__global__ void k_deg(const int* __restrict__ col, float* __restrict__ degf, int n) {
  int e = blockIdx.x * blockDim.x + threadIdx.x;
  if (e < n) atomicAdd(&degf[col[e]], 1.0f);
}
__global__ void k_dinv(const float* __restrict__ degf, float* __restrict__ dinv, int n) {
  int i = blockIdx.x * blockDim.x + threadIdx.x;
  if (i < n) { float d = degf[i]; dinv[i] = (d > 0.0f) ? rsqrtf(d) : 0.0f; }
}
__global__ void k_coeff(const int* __restrict__ row, const int* __restrict__ col,
                        const float* __restrict__ attr, const float* __restrict__ dinv,
                        float* __restrict__ coeff, int n) {
  int e = blockIdx.x * blockDim.x + threadIdx.x;
  if (e < n) coeff[e] = attr[e] * dinv[row[e]] * dinv[col[e]];
}

// ----------------------------------------------------- bucket edges by type
__global__ void k_bucket(const int* __restrict__ etype, unsigned* __restrict__ counts,
                         int* __restrict__ bucket, int n) {
  int e = blockIdx.x * blockDim.x + threadIdx.x;
  if (e < n) {
    int t = etype[e] & 3;
    unsigned p = atomicAdd(&counts[t], 1u);
    bucket[(size_t)t * E_NODES + p] = e;
  }
}

// ---- pre-convert all 3 weight tensors to bf16, WMMA-B-fragment-major layout
// value at (set, r, s, t, lane, j) = W_set[r][32*s + kmap(j,lane)][16*t + (lane&15)]
__global__ void k_wfrag(const float* __restrict__ w1, const float* __restrict__ w2,
                        const float* __restrict__ w3, __bf16* __restrict__ wfrag) {
  int tid = blockIdx.x * blockDim.x + threadIdx.x;
  if (tid >= 3 * WFRAG_SET) return;
  int set = tid >> 16;
  int idx = tid & (WFRAG_SET - 1);
  int j = idx & 15;
  int l = (idx >> 4) & 31;
  int t = (idx >> 9) & 7;
  int s = (idx >> 12) & 3;
  int r = (idx >> 14) & 3;
  int k = 32 * s + kmap(j, l);
  int n = 16 * t + (l & 15);
  const float* w = (set == 0) ? w1 : ((set == 1) ? w2 : w3);
  wfrag[tid] = (__bf16)w[(r * D_DIM + k) * D_DIM + n];
}

// -------- core: gathered 16-edge x 128 x 128 GEMM via WMMA + atomic scatter
__global__ void __launch_bounds__(32)
k_gemm_scatter(const __bf16* __restrict__ xbf, const __bf16* __restrict__ wfragL,
               const int* __restrict__ bucket, const unsigned* __restrict__ counts,
               const int* __restrict__ rowidx, const int* __restrict__ colidx,
               const float* __restrict__ coeff, float* __restrict__ zacc) {
  __shared__ __align__(16) __bf16 sA[16 * D_DIM];  // 16 gathered source rows, bf16
  __shared__ int   sCol[16];
  __shared__ float sC[16];

  const int r = blockIdx.y;
  const int cnt = (int)counts[r];
  const int tile0 = blockIdx.x * 16;
  if (tile0 >= cnt) return;
  const int L = threadIdx.x;

  // gather: lane pair (L>>1) loads one 128-elem bf16 row (64 elems per lane)
  {
    int m2 = L >> 1;
    int co = (L & 1) * 64;
    int eidx = tile0 + m2;
    int e = (eidx < cnt) ? bucket[(size_t)r * E_NODES + eidx] : -1;
    size_t src = (e >= 0) ? (size_t)rowidx[e] : 0;
    const uint4* sp = (const uint4*)(xbf + src * D_DIM + co);
    uint4* dp = (uint4*)(sA + m2 * D_DIM + co);
#pragma unroll
    for (int c = 0; c < 8; ++c) dp[c] = sp[c];
  }
  if (L < 16) {
    int eidx = tile0 + L;
    if (eidx < cnt) {
      int e = bucket[(size_t)r * E_NODES + eidx];
      sCol[L] = colidx[e];
      sC[L]   = coeff[e];
    } else { sCol[L] = 0; sC[L] = 0.0f; }
  }
  __syncthreads();

  v8f acc[8] = {};
  const __bf16* wb = wfragL + (size_t)r * WFRAG_REL;
  const int m  = L & 15;
  const int kb = (L < 16) ? 0 : 8;

#pragma unroll
  for (int s = 0; s < 4; ++s) {          // K steps of 32
    Frag16 fa;
    const __bf16* ap = sA + m * D_DIM + 32 * s + kb;
    fa.u[0] = *(const uint4*)(ap);       // elems j=0..7  -> k = base + j
    fa.u[1] = *(const uint4*)(ap + 16);  // elems j=8..15 -> k = base + 8 + j
#pragma unroll
    for (int t = 0; t < 8; ++t) {        // 8 N-subtiles of 16 (N=128)
      Frag16 fb;
      const uint4* wp = (const uint4*)(wb + ((size_t)((s * 8 + t) * 32 + L)) * 16);
      fb.u[0] = wp[0];
      fb.u[1] = wp[1];
      acc[t] = __builtin_amdgcn_wmma_f32_16x16x32_bf16(
          false, fa.v, false, fb.v, (short)0, acc[t], false, false);
    }
  }

  // scatter: C/D layout — lane L holds rows v + 8*(L>>4), col (L&15) of each subtile
  const int hh = L >> 4, nloc = L & 15;
#pragma unroll
  for (int t = 0; t < 8; ++t) {
#pragma unroll
    for (int v = 0; v < 8; ++v) {
      int mm = v + 8 * hh;
      float c = sC[mm];
      if (c != 0.0f)
        atomicAdd(&zacc[(size_t)sCol[mm] * D_DIM + t * 16 + nloc], acc[t][v] * c);
    }
  }
}

// ------------------------------------------- bias + leakyReLU + accumulate
__global__ void k_finalize(const float* __restrict__ zacc, const float* __restrict__ bias,
                           float* __restrict__ out, __bf16* __restrict__ xbf, int n) {
  int i = blockIdx.x * blockDim.x + threadIdx.x;
  if (i < n) {
    float v = zacc[i] + bias[i & (D_DIM - 1)];
    v = (v > 0.0f) ? v : v * NEG_SLOPE;
    out[i] += v;            // running (x + z1 + z2 + z3)
    xbf[i] = (__bf16)v;     // next layer's input, bf16
  }
}
__global__ void k_scale(float* out, int n) {
  int i = blockIdx.x * blockDim.x + threadIdx.x;
  if (i < n) out[i] *= 0.25f;
}

// ---------------------------------------------------------------------------
extern "C" void kernel_launch(void* const* d_in, const int* in_sizes, int n_in,
                              void* d_out, int out_size, void* d_ws, size_t ws_size,
                              hipStream_t stream) {
  const int* ei    = (const int*)d_in[0];       // [2,E]
  const int* row   = ei;
  const int* col   = ei + E_NODES;
  const int* etype = (const int*)d_in[1];
  const float* attr = (const float*)d_in[2];
  const float* w1 = (const float*)d_in[3];
  const float* b1 = (const float*)d_in[4];
  const float* w2 = (const float*)d_in[5];
  const float* b2 = (const float*)d_in[6];
  const float* w3 = (const float*)d_in[7];
  const float* b3 = (const float*)d_in[8];
  float* out = (float*)d_out;

  // --- workspace layout (256B aligned slabs) ---
  char* ws = (char*)d_ws;
  size_t off = 0;
  auto alloc = [&](size_t bytes) -> char* {
    char* p = ws + off;
    off = (off + bytes + 255) & ~(size_t)255;
    return p;
  };
  float*    degf   = (float*)alloc(sizeof(float) * E_NODES);
  float*    dinv   = (float*)alloc(sizeof(float) * E_NODES);
  float*    coeff  = (float*)alloc(sizeof(float) * E_NODES);
  unsigned* counts = (unsigned*)alloc(sizeof(unsigned) * 8);
  int*      bucket = (int*)alloc(sizeof(int) * (size_t)R_REL * E_NODES);
  __bf16*   wfrag  = (__bf16*)alloc(sizeof(__bf16) * 3 * WFRAG_SET);
  __bf16*   xbf    = (__bf16*)alloc(sizeof(__bf16) * (size_t)E_NODES * D_DIM);
  float*    zacc   = (float*)alloc(sizeof(float) * (size_t)E_NODES * D_DIM);
  (void)ws_size; (void)in_sizes; (void)n_in; (void)out_size;

  const int ND = E_NODES * D_DIM;
  const int TB = 256;
  const int gE  = (E_NODES + TB - 1) / TB;
  const int gND = (ND + TB - 1) / TB;

  // degree norm + per-edge coefficient (shared across the 3 layers)
  k_zero_f32<<<gE, TB, 0, stream>>>(degf, E_NODES);
  k_zero_u32<<<1, 64, 0, stream>>>(counts, 8);
  k_deg<<<gE, TB, 0, stream>>>(col, degf, E_NODES);
  k_dinv<<<gE, TB, 0, stream>>>(degf, dinv, E_NODES);
  k_coeff<<<gE, TB, 0, stream>>>(row, col, attr, dinv, coeff, E_NODES);

  // bucket edges by relation; pre-convert weights to bf16 fragment layout
  k_bucket<<<gE, TB, 0, stream>>>(etype, counts, bucket, E_NODES);
  k_wfrag<<<(3 * WFRAG_SET + TB - 1) / TB, TB, 0, stream>>>(w1, w2, w3, wfrag);

  // x = ones; out accumulates x + z1 + z2 + z3
  k_init_out_x<<<gND, TB, 0, stream>>>(out, xbf, ND);

  const float* biases[3] = { b1, b2, b3 };
  dim3 ggrid(E_NODES / 16, R_REL);   // 12500 x 4 tiles, 1 wave32 per block
  for (int layer = 0; layer < 3; ++layer) {
    k_zero_f32<<<gND, TB, 0, stream>>>(zacc, ND);
    k_gemm_scatter<<<ggrid, 32, 0, stream>>>(xbf, wfrag + (size_t)layer * WFRAG_SET,
                                             bucket, counts, row, col, coeff, zacc);
    k_finalize<<<gND, TB, 0, stream>>>(zacc, biases[layer], out, xbf, ND);
  }
  k_scale<<<gND, TB, 0, stream>>>(out, ND);
}
